// WindowAttention_74569222193820
// MI455X (gfx1250) — matrix operand order repaired and checked by reading, not compiled, not run on hardware
//
#include <hip/hip_runtime.h>
#include <hip/hip_bf16.h>

// ---- CDNA5 WMMA types (wave32) ----
typedef __bf16 bf16_t;
typedef __attribute__((ext_vector_type(16))) __bf16 v16bf;
typedef __attribute__((ext_vector_type(8)))  __bf16 v8bf;
typedef __attribute__((ext_vector_type(8)))  float  v8f;

#define NTOK 49      // window tokens
#define CDIM 128     // channels
#define NPAD 64      // tokens padded to WMMA multiple
#define QLD  136     // bf16 row stride for q/kv tiles (272B: 16B-aligned, bank-spread)
#define KVTLD 80     // bf16 row stride for kv^T (160B)
#define PLD  80      // bf16 row stride for probabilities
#define OTLD 52      // f32 row stride for transposed output staging

// LDS byte offsets (phase-overlaid)
#define OFF_Q   0            // [64][136] bf16 = 17408 B   (phase 1-2; overlaid by P then OT)
#define OFF_KV  17408        // [64][136] bf16 = 17408 B   (phase 1-2)
#define OFF_KVT 34816        // [128][80] bf16 = 20480 B   (phase 1-4)
#define SMEM_BYTES 55296

static __device__ inline bf16_t f2bf(float f) {
  unsigned u = __float_as_uint(f);
  u += 0x7fffu + ((u >> 16) & 1u);               // round-to-nearest-even
  unsigned short h = (unsigned short)(u >> 16);
  return __builtin_bit_cast(bf16_t, h);
}

// A-matrix tile (16xK slice, 16-bit): lane = M (0..15), half-wave selects K sub-blocks.
// VGPR 0-3: K = h*8 + 0..7 ; VGPR 4-7: K = 16 + h*8 + 0..7   (two 16B LDS loads)
static __device__ inline v16bf load_tile_a(const bf16_t* base, int ld, int k0, int lane) {
  int m = lane & 15, h = (lane >> 4) & 1;
  const bf16_t* p = base + m * ld + k0 + h * 8;
  v8bf lo = *(const v8bf*)p;
  v8bf hi = *(const v8bf*)(p + 16);
  v16bf r;
#pragma unroll
  for (int i = 0; i < 8; ++i) { r[i] = lo[i]; r[i + 8] = hi[i]; }
  return r;
}

// B-matrix tile (Kx16, 16-bit): lane = N (0..15), half-wave selects K sub-blocks
// (VGPR 0-3: K = h*8 + 0..7 ; VGPR 4-7: K = 16 + h*8 + 0..7 -> two 16B LDS loads)
static __device__ inline v16bf load_tile_b(const bf16_t* base, int ld, int k0, int lane) {
  int n = lane & 15, h = (lane >> 4) & 1;
  const bf16_t* p = base + n * ld + k0 + h * 8;
  v8bf lo = *(const v8bf*)p;
  v8bf hi = *(const v8bf*)(p + 16);
  v16bf r;
#pragma unroll
  for (int i = 0; i < 8; ++i) { r[i] = lo[i]; r[i + 8] = hi[i]; }
  return r;
}

__global__ __launch_bounds__(128) void WindowAttention_74569222193820_kernel(
    const float* __restrict__ qg, const float* __restrict__ kvg,
    const float* __restrict__ maskg, float* __restrict__ outg) {
  __shared__ __align__(16) unsigned char smem[SMEM_BYTES];
  bf16_t* q_s   = (bf16_t*)(smem + OFF_Q);     // [64][QLD]
  bf16_t* kv_s  = (bf16_t*)(smem + OFF_KV);    // [64][QLD]
  bf16_t* kvT_s = (bf16_t*)(smem + OFF_KVT);   // [128][KVTLD]
  bf16_t* p_s   = (bf16_t*)(smem + OFF_Q);     // [64][PLD]  (overlays dead q)
  float*  ot_s  = (float*)(smem + OFF_Q);      // [128][OTLD] (overlays everything, last phase)

  const int b    = blockIdx.x;
  const int tid  = threadIdx.x;
  const int lane = tid & 31;
  const int wave = tid >> 5;
  const int m0   = wave * 16;                  // this wave's 16-row strip
  const int colb = lane & 15;
  const int rowo = ((lane >> 4) & 1) * 8;      // C/D layout: lanes16-31 hold M=8..15

  // ---- phase 0: zero the token-padding (rows/cols 49..63) ----
  for (int i = tid; i < 15 * CDIM; i += 128) {
    int r = NTOK + i / CDIM, c = i % CDIM;
    q_s [r * QLD + c]   = f2bf(0.0f);
    kv_s[r * QLD + c]   = f2bf(0.0f);
    kvT_s[c * KVTLD + r] = f2bf(0.0f);
  }

  // ---- phase 1: global fp32 -> LDS bf16 (128-bit loads), build kv^T ----
  const float4* qv  = (const float4*)(qg  + (size_t)b * NTOK * CDIM);
  const float4* kvv = (const float4*)(kvg + (size_t)b * NTOK * CDIM);
  for (int idx = tid; idx < (NTOK * CDIM) / 4; idx += 128) {
    float4 a = qv[idx];
    float4 k = kvv[idx];
    int r = (idx * 4) / CDIM, c = (idx * 4) % CDIM;
    q_s[r * QLD + c + 0] = f2bf(a.x); q_s[r * QLD + c + 1] = f2bf(a.y);
    q_s[r * QLD + c + 2] = f2bf(a.z); q_s[r * QLD + c + 3] = f2bf(a.w);
    kv_s[r * QLD + c + 0] = f2bf(k.x); kv_s[r * QLD + c + 1] = f2bf(k.y);
    kv_s[r * QLD + c + 2] = f2bf(k.z); kv_s[r * QLD + c + 3] = f2bf(k.w);
    kvT_s[(c + 0) * KVTLD + r] = f2bf(k.x); kvT_s[(c + 1) * KVTLD + r] = f2bf(k.y);
    kvT_s[(c + 2) * KVTLD + r] = f2bf(k.z); kvT_s[(c + 3) * KVTLD + r] = f2bf(k.w);
  }
  __syncthreads();

  // ---- phase 2: S = q . kv^T   (16x64 strip per wave, K=128) ----
  v8f acc[4];
#pragma unroll
  for (int nt = 0; nt < 4; ++nt) acc[nt] = (v8f){0.f,0.f,0.f,0.f,0.f,0.f,0.f,0.f};
#pragma unroll
  for (int kt = 0; kt < 4; ++kt) {
    v16bf a = load_tile_a(q_s + m0 * QLD, QLD, kt * 32, lane);
#pragma unroll
    for (int nt = 0; nt < 4; ++nt) {
      v16bf bm = load_tile_b(kv_s + nt * 16 * QLD, QLD, kt * 32, lane);
      acc[nt] = __builtin_amdgcn_wmma_f32_16x16x32_bf16(
          false, a, false, bm, (short)0, acc[nt], false, false);
    }
  }
  __syncthreads();   // all q reads done before P overlays q

  // ---- phase 3: mask add + row softmax in registers (wave32 half-wave shuffles) ----
  const float* mp = maskg + (size_t)(b & 63) * NTOK * NTOK;  // window id = b % nW
#pragma unroll
  for (int nt = 0; nt < 4; ++nt) {
#pragma unroll
    for (int r = 0; r < 8; ++r) {
      int row = m0 + rowo + r;
      int col = nt * 16 + colb;
      float v = acc[nt][r];
      if (col < NTOK) { if (row < NTOK) v += mp[row * NTOK + col]; }
      else v = -1.0e30f;                     // padded keys vanish in softmax
      acc[nt][r] = v;
    }
  }
#pragma unroll
  for (int r = 0; r < 8; ++r) {
    float mrow = -3.0e38f;
#pragma unroll
    for (int nt = 0; nt < 4; ++nt) mrow = fmaxf(mrow, acc[nt][r]);
    mrow = fmaxf(mrow, __shfl_xor(mrow, 1));
    mrow = fmaxf(mrow, __shfl_xor(mrow, 2));
    mrow = fmaxf(mrow, __shfl_xor(mrow, 4));
    mrow = fmaxf(mrow, __shfl_xor(mrow, 8));
    float e[4], ssum = 0.f;
#pragma unroll
    for (int nt = 0; nt < 4; ++nt) { e[nt] = __expf(acc[nt][r] - mrow); ssum += e[nt]; }
    ssum += __shfl_xor(ssum, 1);
    ssum += __shfl_xor(ssum, 2);
    ssum += __shfl_xor(ssum, 4);
    ssum += __shfl_xor(ssum, 8);
    float inv = 1.0f / ssum;
    int row = m0 + rowo + r;
#pragma unroll
    for (int nt = 0; nt < 4; ++nt)
      p_s[row * PLD + nt * 16 + colb] = f2bf(e[nt] * inv);
  }
  __syncthreads();   // P fully written

  // ---- phase 4: O = P . kv   (16x128 strip per wave, K=64) ----
  v8f o[8];
#pragma unroll
  for (int ct = 0; ct < 8; ++ct) o[ct] = (v8f){0.f,0.f,0.f,0.f,0.f,0.f,0.f,0.f};
#pragma unroll
  for (int kt = 0; kt < 2; ++kt) {
    v16bf a = load_tile_a(p_s + m0 * PLD, PLD, kt * 32, lane);
#pragma unroll
    for (int ct = 0; ct < 8; ++ct) {
      v16bf bm = load_tile_b(kvT_s + ct * 16 * KVTLD, KVTLD, kt * 32, lane);
      o[ct] = __builtin_amdgcn_wmma_f32_16x16x32_bf16(
          false, a, false, bm, (short)0, o[ct], false, false);
    }
  }
  __syncthreads();   // P / kvT dead, OT overlays

  // ---- phase 5: stage O transposed ([c][n]) so global writes are coalesced ----
#pragma unroll
  for (int ct = 0; ct < 8; ++ct) {
#pragma unroll
    for (int r = 0; r < 8; ++r) {
      int row = m0 + rowo + r;               // token n
      int c   = ct * 16 + colb;              // channel
      if (row < NTOK) ot_s[c * OTLD + row] = o[ct][r];
    }
  }
  __syncthreads();

  float* op = outg + (size_t)b * NTOK * CDIM;
  for (int i = tid; i < NTOK * CDIM; i += 128) {
    int c = i / NTOK, n = i % NTOK;          // out flat index = c*49 + n
    op[i] = ot_s[c * OTLD + n];
  }
}

extern "C" void kernel_launch(void* const* d_in, const int* in_sizes, int n_in,
                              void* d_out, int out_size, void* d_ws, size_t ws_size,
                              hipStream_t stream) {
  const float* q    = (const float*)d_in[0];
  const float* kv   = (const float*)d_in[1];
  const float* mask = (const float*)d_in[2];
  float* out = (float*)d_out;
  int B = in_sizes[0] / (NTOK * CDIM);   // 8192
  if (B <= 0) return;
  WindowAttention_74569222193820_kernel<<<B, 128, 0, stream>>>(q, kv, mask, out);
}
